// KernelOptimizedAttention_46265387712879
// MI455X (gfx1250) — matrix operand order repaired and checked by reading, not compiled, not run on hardware
//
#include <hip/hip_runtime.h>

// ---------------------------------------------------------------------------
// CDNA5 (gfx1250) GQA attention layer:
//   Q/K/V projections -> flash attention (online softmax) -> O projection
// Matmuls via v_wmma_f32_16x16x32_f16 (wave32), f32 accumulation.
// Flash-attention Q/K tiles staged into LDS by the Tensor Data Mover
// (tensor_load_to_lds, TENSORcnt-tracked).
// ---------------------------------------------------------------------------

typedef __attribute__((ext_vector_type(16))) _Float16 v16h;
typedef __attribute__((ext_vector_type(8)))  _Float16 h8;
typedef __attribute__((ext_vector_type(8)))  float    v8f;
typedef __attribute__((ext_vector_type(4)))  unsigned int u32x4;
typedef __attribute__((ext_vector_type(8)))  int i32x8;
typedef __attribute__((ext_vector_type(4)))  int i32x4;

// D = A(16x32 f16) * B(32x16 f16) + C(16x16 f32)
__device__ inline v8f wmma_f16(v16h a, v16h b, v8f c) {
    return __builtin_amdgcn_wmma_f32_16x16x32_f16(
        /*neg_a=*/false, a, /*neg_b=*/false, b,
        /*c_mod=*/(short)0, c, /*reuse_a=*/false, /*reuse_b=*/false);
}

// A-fragment (16x32 f16) from row-major LDS tile [m][k], row stride `ld` halves.
// ISA layout: lane L (m = L%16, hi = L/16); element e: K = (e/8)*16 + hi*8 + e%8
__device__ inline v16h ld_a_frag(const _Float16* base, int ld, int m0, int koff, int lane) {
    const int hi = lane >> 4, mm = lane & 15;
    const _Float16* p = base + (size_t)(m0 + mm) * ld + koff + hi * 8;
    h8 lo = *(const h8*)(p);
    h8 hh = *(const h8*)(p + 16);
    v16h f;
#pragma unroll
    for (int i = 0; i < 8; ++i) { f[i] = lo[i]; f[i + 8] = hh[i]; }
    return f;
}

// B-fragment (32x16 f16) from *transposed* LDS tile Bt[n][k], row stride `ld`.
// ISA layout: lane L (n = L%16, hi = L/16); element e: K = (e/8)*16 + hi*8 + e%8
__device__ inline v16h ld_bt_frag(const _Float16* base, int ld, int n0, int koff, int lane) {
    const int hi = lane >> 4, nn = lane & 15;
    const _Float16* p = base + (size_t)(n0 + nn) * ld + koff + hi * 8;
    h8 lo = *(const h8*)(p);
    h8 hh = *(const h8*)(p + 16);
    v16h f;
#pragma unroll
    for (int i = 0; i < 8; ++i) { f[i] = lo[i]; f[i + 8] = hh[i]; }
    return f;
}

// ---------------------------------------------------------------------------
// Tensor Data Mover: load a 2D f16 tile (128 x 128 elements) from a row-major
// f16 tensor (row length `stride_elems`) at `gptr` into LDS offset `lds_off`.
// D# packing per CDNA5 ISA ch. 8 (group0: count/lds/global/type, group1:
// data_size, tensor dims, tile dims, dim0 stride).
// ---------------------------------------------------------------------------
__device__ inline void tdm_load_2d_f16(unsigned lds_off, const void* gptr,
                                       unsigned tensor_w, unsigned tensor_h,
                                       unsigned stride_elems) {
    unsigned long long ga = (unsigned long long)gptr;
    u32x4 g0;
    g0[0] = 1u;                                   // count=1, user mode
    g0[1] = lds_off;                              // lds_addr (bytes)
    g0[2] = (unsigned)(ga & 0xffffffffu);         // global_addr[31:0]
    g0[3] = (unsigned)((ga >> 32) & 0x01ffffffu)  // global_addr[56:32]
          | (2u << 30);                           // type=2 ("image")
    i32x8 g1;
    g1[0] = (int)(1u << 16);                                  // data_size=2B
    g1[1] = (int)((tensor_w & 0xffffu) << 16);                // tensor_dim0[15:0]
    g1[2] = (int)(((tensor_w >> 16) & 0xffffu)                // tensor_dim0[31:16]
          |       ((tensor_h & 0xffffu) << 16));              // tensor_dim1[15:0]
    g1[3] = (int)(((tensor_h >> 16) & 0xffffu)                // tensor_dim1[31:16]
          |       (128u << 16));                              // tile_dim0 = 128
    g1[4] = (int)128u;                                        // tile_dim1 = 128
    g1[5] = (int)stride_elems;                                // dim0_stride[31:0]
    g1[6] = 0;                                                // dim0_stride[47:32], dim1_stride lo
    g1[7] = 0;
    i32x4 z4 = {0, 0, 0, 0};
#if __clang_major__ >= 23
    i32x8 z8 = {0, 0, 0, 0, 0, 0, 0, 0};
    __builtin_amdgcn_tensor_load_to_lds(g0, g1, z4, z4, z8, 0);
#else
    __builtin_amdgcn_tensor_load_to_lds(g0, g1, z4, z4, 0);
#endif
}

// ---------------------------------------------------------------------------
// Generic WMMA GEMM:  C[M,N] = A[M,K] * B[K,N] + bias[N]
// A: f32 (converted on load) or f16;  B: f32 weights;  C: f16 or f32.
// Block tile 128x128, BK=32, 256 threads (8 waves).
// ---------------------------------------------------------------------------
template <bool A_HALF, bool OUT_HALF>
__global__ __launch_bounds__(256) void wmma_gemm_bias(
    const void* __restrict__ Ap, const float* __restrict__ Bp,
    const float* __restrict__ bias, void* __restrict__ Cp,
    int M, int N, int K, int lda, int ldb, int ldc)
{
    __shared__ _Float16 As[128 * 32];   // [row][k]      8 KB
    __shared__ _Float16 Bts[128 * 32];  // [n][k] transp 8 KB

    const int tid = threadIdx.x, lane = tid & 31, w = tid >> 5;
    const int bm = blockIdx.y * 128, bn = blockIdx.x * 128;

    v8f acc[8] = {};

    const int arow = tid >> 1, acolh = (tid & 1) * 16;  // A: 16 halves/thread
    const int bk = tid & 31,  bn0 = (tid >> 5) * 16;    // B: 16 floats/thread

    for (int k0 = 0; k0 < K; k0 += 32) {
        // ---- stage A tile (128x32) into LDS as f16 ----
        if (A_HALF) {
            const _Float16* Ag = (const _Float16*)Ap + (size_t)(bm + arow) * lda + k0 + acolh;
            _Float16* d = As + arow * 32 + acolh;
            *(h8*)(d)     = *(const h8*)(Ag);
            *(h8*)(d + 8) = *(const h8*)(Ag + 8);
        } else {
            const float* Ag = (const float*)Ap + (size_t)(bm + arow) * lda + k0 + acolh;
            const float4* A4 = (const float4*)Ag;
            float4 f0 = A4[0], f1 = A4[1], f2 = A4[2], f3 = A4[3];
            float tf[16] = { f0.x, f0.y, f0.z, f0.w, f1.x, f1.y, f1.z, f1.w,
                             f2.x, f2.y, f2.z, f2.w, f3.x, f3.y, f3.z, f3.w };
            _Float16* d = As + arow * 32 + acolh;
#pragma unroll
            for (int j = 0; j < 16; ++j) d[j] = (_Float16)tf[j];
        }
        // ---- stage B tile (32x128) transposed into LDS as f16 ----
        {
            const float* Bg = Bp + (size_t)(k0 + bk) * ldb + bn + bn0;
            const float4* B4 = (const float4*)Bg;
            float4 g0 = B4[0], g1 = B4[1], g2 = B4[2], g3 = B4[3];
            float tb[16] = { g0.x, g0.y, g0.z, g0.w, g1.x, g1.y, g1.z, g1.w,
                             g2.x, g2.y, g2.z, g2.w, g3.x, g3.y, g3.z, g3.w };
#pragma unroll
            for (int j = 0; j < 16; ++j) Bts[(bn0 + j) * 32 + bk] = (_Float16)tb[j];
        }
        // prefetch next B slab (global_prefetch_b8)
        if (k0 + 32 < K) {
            const float* nb = Bp + (size_t)(k0 + 32 + bk) * ldb + bn + bn0;
            __builtin_prefetch(nb, 0, 0);
        }
        __syncthreads();

        v16h af = ld_a_frag(As, 32, w * 16, 0, lane);
#pragma unroll
        for (int nt = 0; nt < 8; ++nt) {
            v16h bf = ld_bt_frag(Bts, 32, nt * 16, 0, lane);
            acc[nt] = wmma_f16(af, bf, acc[nt]);
        }
        __syncthreads();
    }

    // ---- epilogue: C layout row = r + hi*8, col = lane%16 ----
    const int hi = lane >> 4, ln = lane & 15;
#pragma unroll
    for (int nt = 0; nt < 8; ++nt) {
        const int col = bn + nt * 16 + ln;
        const float bvv = bias[col];
#pragma unroll
        for (int r = 0; r < 8; ++r) {
            const int row = bm + w * 16 + hi * 8 + r;
            float v = acc[nt][r] + bvv;
            if (OUT_HALF) ((_Float16*)Cp)[(size_t)row * ldc + col] = (_Float16)v;
            else          ((float*)Cp)[(size_t)row * ldc + col] = v;
        }
    }
}

// ---------------------------------------------------------------------------
// Flash attention (GQA, n_rep=4): grid = (NH=32, S/128=16), 256 threads.
// Q/K tiles DMA'd into LDS by the TDM; V transposed manually; online softmax.
// Qh/Ah: [S, 4096] f16.  Kh/Vh: [S, 1024] f16.  mask: [S,S] f32 (added).
// ---------------------------------------------------------------------------
__global__ __launch_bounds__(256) void flash_attn_wmma(
    const _Float16* __restrict__ Qh, const _Float16* __restrict__ Kh,
    const _Float16* __restrict__ Vh, const float* __restrict__ mask,
    _Float16* __restrict__ Ah)
{
    extern __shared__ _Float16 smem[];
    _Float16* Qs = smem;                  // [qrow][d]  128x128  32 KB  (LDS off 0)
    _Float16* Ks = smem + 128 * 128;      // [krow][d]  128x128  32 KB  (LDS off 32768)
    _Float16* Vt = smem + 2 * 128 * 128;  // [d][krow]  128x128  32 KB
    _Float16* Ps = smem + 3 * 128 * 128;  // 8 waves x [16][128]  32 KB

    const int tid = threadIdx.x, lane = tid & 31, w = tid >> 5;
    const int h = blockIdx.x, qb = blockIdx.y, kvh = h >> 2;
    const int hi = lane >> 4, ln = lane & 15;

    // ---- Q block via TDM (issued once, by wave 0) ----
    if (w == 0) {
        tdm_load_2d_f16(/*lds_off=*/0u,
                        Qh + (size_t)(qb * 128) * 4096 + h * 128,
                        /*tensor_w=*/4096u, /*tensor_h=*/2048u,
                        /*stride=*/4096u);
    }

    v8f o[8] = {};
    float mrun[8], lrun[8];
#pragma unroll
    for (int r = 0; r < 8; ++r) { mrun[r] = -1e30f; lrun[r] = 0.f; }

    const float scale = 0.08838834764831845f;  // 128^-0.5

    for (int kb = 0; kb < 16; ++kb) {
        // ---- K block via TDM ----
        if (w == 0) {
            tdm_load_2d_f16(/*lds_off=*/(unsigned)(128 * 128 * sizeof(_Float16)),
                            Kh + (size_t)(kb * 128) * 1024 + kvh * 128,
                            /*tensor_w=*/1024u, /*tensor_h=*/2048u,
                            /*stride=*/1024u);
        }
        // ---- V block transposed (manual; TDM cannot transpose) ----
        {
            const int r = tid >> 1, db = (tid & 1) * 64;
            const _Float16* vsrc = Vh + (size_t)(kb * 128 + r) * 1024 + kvh * 128 + db;
            h8 vv[8];
#pragma unroll
            for (int i = 0; i < 8; ++i) vv[i] = ((const h8*)vsrc)[i];
#pragma unroll
            for (int i = 0; i < 8; ++i) {
#pragma unroll
                for (int j = 0; j < 8; ++j)
                    Vt[(size_t)(db + i * 8 + j) * 128 + r] = vv[i][j];
            }
        }
        // TDM completion (in-order per wave; covers the Q load on kb==0)
        if (w == 0) __builtin_amdgcn_s_wait_tensorcnt(0);
        __syncthreads();

        // ---- scores S = Q * K^T for this wave's 16 rows x 128 keys ----
        v8f s[8] = {};
#pragma unroll
        for (int ks = 0; ks < 4; ++ks) {
            v16h af = ld_a_frag(Qs, 128, w * 16, ks * 32, lane);
#pragma unroll
            for (int nt = 0; nt < 8; ++nt) {
                v16h bf = ld_bt_frag(Ks, 128, nt * 16, ks * 32, lane);
                s[nt] = wmma_f16(af, bf, s[nt]);
            }
        }

        // ---- scale + additive mask ----
        const int q0 = qb * 128 + w * 16 + hi * 8;
        const int kc = kb * 128 + ln;
#pragma unroll
        for (int nt = 0; nt < 8; ++nt) {
#pragma unroll
            for (int r = 0; r < 8; ++r)
                s[nt][r] = s[nt][r] * scale + mask[(size_t)(q0 + r) * 2048 + kc + nt * 16];
        }

        // ---- online softmax update (row reductions across 16 lanes) ----
#pragma unroll
        for (int r = 0; r < 8; ++r) {
            float mx = s[0][r];
#pragma unroll
            for (int nt = 1; nt < 8; ++nt) mx = fmaxf(mx, s[nt][r]);
#pragma unroll
            for (int off = 8; off >= 1; off >>= 1)
                mx = fmaxf(mx, __shfl_xor(mx, off, 32));
            const float mnew  = fmaxf(mrun[r], mx);
            const float alpha = __expf(mrun[r] - mnew);
            lrun[r] *= alpha;
#pragma unroll
            for (int t = 0; t < 8; ++t) o[t][r] *= alpha;
            float rs = 0.f;
#pragma unroll
            for (int nt = 0; nt < 8; ++nt) {
                float p = __expf(s[nt][r] - mnew);
                s[nt][r] = p;
                rs += p;
            }
#pragma unroll
            for (int off = 8; off >= 1; off >>= 1)
                rs += __shfl_xor(rs, off, 32);
            lrun[r] += rs;
            mrun[r] = mnew;
        }

        // ---- write P (f16) into this wave's LDS strip in A-tile layout ----
        _Float16* Pw = Ps + w * (16 * 128);
#pragma unroll
        for (int nt = 0; nt < 8; ++nt) {
#pragma unroll
            for (int r = 0; r < 8; ++r)
                Pw[(hi * 8 + r) * 128 + nt * 16 + ln] = (_Float16)s[nt][r];
        }
        // wave-local LDS RAW fence (CDNA5 split counter)
        asm volatile("s_wait_dscnt 0" ::: "memory");

        // ---- O += P * V ----
#pragma unroll
        for (int ks = 0; ks < 4; ++ks) {
            v16h af = ld_a_frag(Pw, 128, 0, ks * 32, lane);
#pragma unroll
            for (int t = 0; t < 8; ++t) {
                v16h bf = ld_bt_frag(Vt, 128, t * 16, ks * 32, lane);
                o[t] = wmma_f16(af, bf, o[t]);
            }
        }
        __syncthreads();
    }

    // ---- normalize and store attention output (f16) ----
#pragma unroll
    for (int t = 0; t < 8; ++t) {
#pragma unroll
        for (int r = 0; r < 8; ++r) {
            const float inv = 1.f / lrun[r];
            Ah[(size_t)(qb * 128 + w * 16 + hi * 8 + r) * 4096 + h * 128 + t * 16 + ln] =
                (_Float16)(o[t][r] * inv);
        }
    }
}

// ---------------------------------------------------------------------------
// Host launcher
// ---------------------------------------------------------------------------
extern "C" void kernel_launch(void* const* d_in, const int* in_sizes, int n_in,
                              void* d_out, int out_size, void* d_ws, size_t ws_size,
                              hipStream_t stream) {
    (void)in_sizes; (void)n_in; (void)out_size; (void)ws_size;
    const float* hidden = (const float*)d_in[0];
    const float* mask   = (const float*)d_in[1];
    const float* Wq = (const float*)d_in[2];
    const float* bq = (const float*)d_in[3];
    const float* Wk = (const float*)d_in[4];
    const float* bk = (const float*)d_in[5];
    const float* Wv = (const float*)d_in[6];
    const float* bv = (const float*)d_in[7];
    const float* Wo = (const float*)d_in[8];
    const float* bo = (const float*)d_in[9];
    float* out = (float*)d_out;

    // workspace layout (40 MB): Qh 16MB | Kh 4MB | Vh 4MB | Ah 16MB (all f16)
    char* ws = (char*)d_ws;
    _Float16* Qh = (_Float16*)(ws);
    _Float16* Kh = (_Float16*)(ws + (size_t)16 * 1024 * 1024);
    _Float16* Vh = (_Float16*)(ws + (size_t)20 * 1024 * 1024);
    _Float16* Ah = (_Float16*)(ws + (size_t)24 * 1024 * 1024);

    dim3 blk(256);
    // Q/K/V projections (f32 in -> f16 out)
    wmma_gemm_bias<false, true><<<dim3(32, 16), blk, 0, stream>>>(
        hidden, Wq, bq, Qh, 2048, 4096, 4096, 4096, 4096, 4096);
    wmma_gemm_bias<false, true><<<dim3(8, 16), blk, 0, stream>>>(
        hidden, Wk, bk, Kh, 2048, 1024, 4096, 4096, 1024, 1024);
    wmma_gemm_bias<false, true><<<dim3(8, 16), blk, 0, stream>>>(
        hidden, Wv, bv, Vh, 2048, 1024, 4096, 4096, 1024, 1024);
    // attention (128 KB dynamic LDS)
    flash_attn_wmma<<<dim3(32, 16), blk, 4 * 128 * 128 * sizeof(_Float16), stream>>>(
        Qh, Kh, Vh, mask, Ah);
    // output projection (f16 in -> f32 out)
    wmma_gemm_bias<true, false><<<dim3(32, 16), blk, 0, stream>>>(
        Ah, Wo, bo, out, 2048, 4096, 4096, 4096, 4096, 4096);
}
